// NTXent9Loss_27178553049670
// MI455X (gfx1250) — matrix operand order repaired and checked by reading, not compiled, not run on hardware
//
#include <hip/hip_runtime.h>

typedef _Float16 v8h  __attribute__((ext_vector_type(8)));
typedef _Float16 v16h __attribute__((ext_vector_type(16)));
typedef float    v8f  __attribute__((ext_vector_type(8)));

union V16 { v16h v; v8h h[2]; };

#define N_TOT   8192          // 2B rows
#define DDIM    256           // feature dim
#define BHALF   4096          // B
#define NT      512           // N_TOT / 16 column tiles
#define NSPLIT  8             // column splits per row-tile
#define TILES_PER_SPLIT (NT / NSPLIT)   // 64
#define SHIFT   2.0f          // sim/TEMP <= 2 exactly (cosine<=1, T=0.5)

// ---------------------------------------------------------------------------
// Phase 1: normalize rows of [proj1;proj2], fold 1/(norm*sqrt(T)) into f16
// rows; write zn row-major AND znT transposed (via LDS staging so the
// transposed stores are 16-byte granules, not 2-byte scatters).
// One wave per row, 8 waves / block.
// ---------------------------------------------------------------------------
__global__ void __launch_bounds__(256) ntxent_normalize(
    const float* __restrict__ p1, const float* __restrict__ p2,
    _Float16* __restrict__ zn, _Float16* __restrict__ znT) {
  __shared__ _Float16 stage[DDIM][8];   // [k][row-in-block], rows are 16B

  const int wave = threadIdx.x >> 5;
  const int lane = threadIdx.x & 31;
  const int row  = blockIdx.x * 8 + wave;

  const float* src = (row < BHALF) ? (p1 + (size_t)row * DDIM)
                                   : (p2 + (size_t)(row - BHALF) * DDIM);
  float4 f0 = ((const float4*)src)[lane * 2 + 0];
  float4 f1 = ((const float4*)src)[lane * 2 + 1];
  float ss = f0.x*f0.x + f0.y*f0.y + f0.z*f0.z + f0.w*f0.w
           + f1.x*f1.x + f1.y*f1.y + f1.z*f1.z + f1.w*f1.w;
#pragma unroll
  for (int m = 16; m >= 1; m >>= 1) ss += __shfl_xor(ss, m, 32);
  float norm  = fmaxf(sqrtf(ss), 1e-8f);
  float scale = 1.41421356237309515f / norm;   // sqrt(2)/norm == 1/(norm*sqrt(T))

  float vals[8] = {f0.x, f0.y, f0.z, f0.w, f1.x, f1.y, f1.z, f1.w};
  v8h h;
#pragma unroll
  for (int j = 0; j < 8; ++j) h[j] = (_Float16)(vals[j] * scale);

  // row-major copy (A-fragment source), fully coalesced
  *(v8h*)(zn + (size_t)row * DDIM + lane * 8) = h;
  // stage for transpose
#pragma unroll
  for (int j = 0; j < 8; ++j) stage[lane * 8 + j][wave] = h[j];
  __syncthreads();

  // thread t owns feature k=t: write 8 rows (16 contiguous bytes) of znT
  const int k    = threadIdx.x;
  const int row0 = blockIdx.x * 8;
  v8h col = *(const v8h*)&stage[k][0];
  *(v8h*)(znT + (size_t)k * N_TOT + row0) = col;
}

// ---------------------------------------------------------------------------
// Phase 2: fused Gram + shifted-sum-exp. Each wave: one 16-row tile (mt) and
// one column split s (64 column tiles). K=256 contracted with 8 chained
// v_wmma_f32_16x16x32_f16 per 16x16 output tile, A fragments resident.
// ---------------------------------------------------------------------------
__global__ void __launch_bounds__(256) ntxent_gram(
    const _Float16* __restrict__ zn, const _Float16* __restrict__ znT,
    float* __restrict__ ws_sums,     // [NT][NSPLIT][16]
    float* __restrict__ ws_pos) {    // [N_TOT]
  const int wave = threadIdx.x >> 5;
  const int lane = threadIdx.x & 31;
  const int flat = blockIdx.x * 8 + wave;          // 0 .. NT*NSPLIT-1
  const int mt   = flat >> 3;                      // row tile (NSPLIT==8)
  const int s    = flat & (NSPLIT - 1);            // column split
  const int m0   = mt * 16;

  // --- resident A fragments: 16x32 f16, ISA layout ---
  // lanes 0-15: M=lane, halves = K {0..7} U {16..23}; lanes 16-31: +8 on K
  const int mrow = m0 + (lane & 15);
  const int koff = (lane < 16) ? 0 : 8;
  V16 a[8];
#pragma unroll
  for (int kc = 0; kc < 8; ++kc) {
    const _Float16* base = zn + (size_t)mrow * DDIM + kc * 32 + koff;
    a[kc].h[0] = *(const v8h*)(base);
    a[kc].h[1] = *(const v8h*)(base + 16);
  }

  float sm[8];
#pragma unroll
  for (int r = 0; r < 8; ++r) sm[r] = 0.0f;
  float posv = 0.0f;

  const int diag_t = mt;
  const int pos_t  = (mt + 256) & (NT - 1);        // +B columns == +256 tiles
  const int hi     = lane >> 4;                    // half-wave id
  const int mysel  = lane & 15;                    // column offset this lane owns
  const int nt0    = s * TILES_PER_SPLIT;

  for (int t = 0; t < TILES_PER_SPLIT; ++t) {
    const int nt = nt0 + t;
    const int n0 = nt * 16;
    v8f c = {};
#pragma unroll
    for (int kc = 0; kc < 8; ++kc) {
      // B 32x16 f16: lane L holds column N=L&15, K in the same order as A (zn row n)
      V16 bu;
      const _Float16* bb = zn + (size_t)(n0 + (lane & 15)) * DDIM + kc * 32 + koff;
      bu.h[0] = *(const v8h*)(bb);
      bu.h[1] = *(const v8h*)(bb + 16);
      v16h b = bu.v;
      c = __builtin_amdgcn_wmma_f32_16x16x32_f16(
              false, a[kc].v, false, b, (short)0, c, false, false);
    }
    const bool isdiag = (nt == diag_t);
    const bool ispos  = (nt == pos_t);
#pragma unroll
    for (int r = 0; r < 8; ++r) {
      // C layout: VGPR r -> row (r + 8*hi), column n0 + (lane&15)
      const float v     = c[r];
      const bool  match = (mysel == r + hi * 8);
      if (ispos && match) posv = v;
      float e = __expf(v - SHIFT);
      if (isdiag && match) e = 0.0f;                // exclude diagonal
      sm[r] += e;
    }
  }

  // reduce within each 16-lane half (halves own disjoint row sets)
#pragma unroll
  for (int r = 0; r < 8; ++r) {
#pragma unroll
    for (int m = 8; m >= 1; m >>= 1) sm[r] += __shfl_xor(sm[r], m, 32);
  }

  float* out = ws_sums + ((size_t)mt * NSPLIT + s) * 16;
  if (lane == 0) {
#pragma unroll
    for (int r = 0; r < 8; ++r) out[r] = sm[r];         // rows 0..7
  }
  if (lane == 16) {
#pragma unroll
    for (int r = 0; r < 8; ++r) out[8 + r] = sm[r];     // rows 8..15
  }

  // positive pair: only the split containing pos_t writes; exactly one lane
  // per row matched (lanes 0-7 -> rows 0-7, lanes 24-31 -> rows 8-15)
  if (pos_t >= nt0 && pos_t < nt0 + TILES_PER_SPLIT) {
    int myrow = -1;
    if (lane < 8)        myrow = lane;
    else if (lane >= 24) myrow = lane - 16;
    if (myrow >= 0) ws_pos[m0 + myrow] = posv;
  }
}

// ---------------------------------------------------------------------------
// Phase 3: combine split partials, lse = SHIFT + log(sum), mean(lse - pos).
// Single block -> fixed reduction order -> deterministic across replays.
// ---------------------------------------------------------------------------
__global__ void __launch_bounds__(1024) ntxent_finalize(
    const float* __restrict__ ws_sums, const float* __restrict__ ws_pos,
    float* __restrict__ out) {
  __shared__ float red[1024];
  float acc = 0.0f;
  for (int row = threadIdx.x; row < N_TOT; row += 1024) {
    const int rt = row >> 4, ri = row & 15;
    float tot = 0.0f;
#pragma unroll
    for (int s2 = 0; s2 < NSPLIT; ++s2)
      tot += ws_sums[((size_t)rt * NSPLIT + s2) * 16 + ri];
    acc += (SHIFT + __logf(tot)) - ws_pos[row];
  }
  red[threadIdx.x] = acc;
  __syncthreads();
  for (int step = 512; step >= 1; step >>= 1) {
    if (threadIdx.x < step) red[threadIdx.x] += red[threadIdx.x + step];
    __syncthreads();
  }
  if (threadIdx.x == 0) out[0] = red[0] * (1.0f / N_TOT);
}

// ---------------------------------------------------------------------------
extern "C" void kernel_launch(void* const* d_in, const int* in_sizes, int n_in,
                              void* d_out, int out_size, void* d_ws, size_t ws_size,
                              hipStream_t stream) {
  (void)in_sizes; (void)n_in; (void)out_size; (void)ws_size;
  const float* p1 = (const float*)d_in[0];
  const float* p2 = (const float*)d_in[1];

  char* ws = (char*)d_ws;
  _Float16* zn      = (_Float16*)ws;                                  // 4 MB
  _Float16* znT     = (_Float16*)(ws + (size_t)N_TOT * DDIM * 2);     // 4 MB
  float*    ws_sums = (float*)(ws + 2 * (size_t)N_TOT * DDIM * 2);    // 256KB area
  float*    ws_pos  = ws_sums + (size_t)NT * NSPLIT * 16;             // 8192 floats

  ntxent_normalize<<<N_TOT / 8, 256, 0, stream>>>(p1, p2, zn, znT);
  ntxent_gram<<<(NT * NSPLIT) / 8, 256, 0, stream>>>(zn, znT, ws_sums, ws_pos);
  ntxent_finalize<<<1, 1024, 0, stream>>>(ws_sums, ws_pos, (float*)d_out);
}
